// XorLayer_33389075759117
// MI455X (gfx1250) — matrix operand order repaired and checked by reading, not compiled, not run on hardware
//
#include <hip/hip_runtime.h>

// XOR-convolution via Walsh-Hadamard transform on gfx1250 WMMA.
// out = H * ((H*p1) .* (H*p2)) / 256,  H[j,n] = (-1)^popc(j&n), H^2 = 256*I.
//
// B-operand (Hadamard chunk) trick: for j = base + e with base = multiple of 16
// and e in [0,16), bits are disjoint, so
//   parity(j & n) = parity(e & (n&15)) XOR parity((base>>4) & (n>>4)).
// The first term is a fixed per-lane 16-elem +/-1 vector (Bpat, built once);
// the second is one bit per chunk -> whole-vector conditional sign flip (xor).

typedef __attribute__((ext_vector_type(16))) _Float16     v16h;
typedef __attribute__((ext_vector_type(8)))  _Float16     v8h;
typedef __attribute__((ext_vector_type(8)))  float        v8f;
typedef __attribute__((ext_vector_type(4)))  float        v4f;
typedef __attribute__((ext_vector_type(8)))  unsigned int v8u;

#define NCLS 256
#define ROWS_PER_WG 16
#define NTHREADS 512   // 16 wave32s; each wave owns one 16-column output tile

// parity of a 4-bit value via LUT constant
__device__ __forceinline__ unsigned parity4(unsigned x) {
  return (0x6996u >> x) & 1u;
}

__global__ __launch_bounds__(NTHREADS)
void xor_conv_wht_kernel(const float* __restrict__ p1,
                         const float* __restrict__ p2,
                         float* __restrict__ out) {
  __shared__ _Float16 sP1[ROWS_PER_WG * NCLS];  // 8 KB
  __shared__ _Float16 sP2[ROWS_PER_WG * NCLS];  // 8 KB
  __shared__ _Float16 sZ [ROWS_PER_WG * NCLS];  // 8 KB

  const int t    = threadIdx.x;
  const int lane = t & 31;
  const int wave = t >> 5;                       // 0..15
  const int row0 = blockIdx.x * ROWS_PER_WG;

  // ---- stage 0: stage both 16x256 f32 tiles into LDS as f16 ----
  {
    const int base = t * 8;                      // 512 threads * 8 = 4096 elems
    const float* g1 = p1 + (size_t)row0 * NCLS + base;
    const float* g2 = p2 + (size_t)row0 * NCLS + base;
    v4f a0 = *(const v4f*)(g1);
    v4f a1 = *(const v4f*)(g1 + 4);
    v4f b0 = *(const v4f*)(g2);
    v4f b1 = *(const v4f*)(g2 + 4);
    v8h h1, h2;
#pragma unroll
    for (int e = 0; e < 4; ++e) {
      h1[e]     = (_Float16)a0[e];
      h1[e + 4] = (_Float16)a1[e];
      h2[e]     = (_Float16)b0[e];
      h2[e + 4] = (_Float16)b1[e];
    }
    *(v8h*)&sP1[base] = h1;
    *(v8h*)&sP2[base] = h2;
  }
  __syncthreads();

  // Operand-layout helpers (wave32, 16x16x32 f16 WMMA; this code's k order: lane half h holds k = 16h..16h+15):
  const int mrow  = lane & 15;                   // A row owned by this lane
  const int abase = (lane < 16) ? 0 : 16;        // A K-group offset
  const int ncol  = wave * 16 + (lane & 15);     // B/D column owned by this lane
  const int ghalf = (lane < 16) ? 0 : 1;         // B K-group (16s) per lane grp
  const int mbase = (lane < 16) ? 0 : 8;         // C/D row base per lane group
  const unsigned nlo = (unsigned)(ncol & 15);
  const unsigned nhi = (unsigned)(ncol >> 4);

  // Per-lane base Hadamard pattern: Bpat[e] = (-1)^parity(e & nlo), built once.
  v8u bpat;
#pragma unroll
  for (int e2 = 0; e2 < 8; ++e2) {
    unsigned plo = parity4((unsigned)(2 * e2)     & nlo) << 15;
    unsigned phi = parity4((unsigned)(2 * e2 + 1) & nlo) << 31;
    bpat[e2] = 0x3C003C00u | plo | phi;          // two packed f16 (+/-1.0)
  }

  // ---- stage 1: Y1 = P1*H, Y2 = P2*H for this wave's 16-column tile ----
  v8f accY1 = {};
  v8f accY2 = {};
#pragma unroll
  for (int kc = 0; kc < 8; ++kc) {
    const int kb = kc * 32;
    // A operands: two contiguous 8-halfword runs per lane (ds_load_b128 each)
    v8h lo1 = *(const v8h*)&sP1[mrow * NCLS + kb + abase];
    v8h hi1 = *(const v8h*)&sP1[mrow * NCLS + kb + abase + 8];
    v8h lo2 = *(const v8h*)&sP2[mrow * NCLS + kb + abase];
    v8h hi2 = *(const v8h*)&sP2[mrow * NCLS + kb + abase + 8];
    v16h A1, A2;
#pragma unroll
    for (int e = 0; e < 8; ++e) {
      A1[e] = lo1[e]; A1[e + 8] = hi1[e];
      A2[e] = lo2[e]; A2[e + 8] = hi2[e];
    }
    // B chunk = Bpat with conditional whole-vector sign flip
    unsigned g    = (unsigned)(2 * kc + ghalf);
    unsigned flip = parity4(g & nhi) * 0x80008000u;
    v16h Bm = __builtin_bit_cast(v16h, bpat ^ flip);

    accY1 = __builtin_amdgcn_wmma_f32_16x16x32_f16(
        false, A1, false, Bm, (short)0, accY1, false, false);
    accY2 = __builtin_amdgcn_wmma_f32_16x16x32_f16(
        false, A2, false, Bm, (short)0, accY2, false, false);
  }

  // ---- pointwise: Z = Y1 .* Y2 / 256, scattered back to LDS as f16 ----
#pragma unroll
  for (int v = 0; v < 8; ++v) {
    float z = accY1[v] * accY2[v] * (1.0f / 256.0f);  // |Z| <= 256: f16-safe
    sZ[(mbase + v) * NCLS + ncol] = (_Float16)z;
  }
  __syncthreads();

  // ---- stage 2: Out = Z*H for this wave's 16-column tile ----
  v8f accO = {};
#pragma unroll
  for (int kc = 0; kc < 8; ++kc) {
    const int kb = kc * 32;
    v8h lo = *(const v8h*)&sZ[mrow * NCLS + kb + abase];
    v8h hi = *(const v8h*)&sZ[mrow * NCLS + kb + abase + 8];
    v16h A;
#pragma unroll
    for (int e = 0; e < 8; ++e) { A[e] = lo[e]; A[e + 8] = hi[e]; }

    unsigned g    = (unsigned)(2 * kc + ghalf);
    unsigned flip = parity4(g & nhi) * 0x80008000u;
    v16h Bm = __builtin_bit_cast(v16h, bpat ^ flip);

    accO = __builtin_amdgcn_wmma_f32_16x16x32_f16(
        false, A, false, Bm, (short)0, accO, false, false);
  }

  // ---- store f32 result; lanes 0-15 / 16-31 each cover 64B coalesced runs ----
#pragma unroll
  for (int v = 0; v < 8; ++v)
    out[(size_t)(row0 + mbase + v) * NCLS + ncol] = accO[v];
}

extern "C" void kernel_launch(void* const* d_in, const int* in_sizes, int n_in,
                              void* d_out, int out_size, void* d_ws, size_t ws_size,
                              hipStream_t stream) {
  (void)n_in; (void)d_ws; (void)ws_size; (void)out_size;
  const float* p1 = (const float*)d_in[0];
  const float* p2 = (const float*)d_in[1];
  float* out = (float*)d_out;
  const int rows = in_sizes[0] / NCLS;            // 2048
  const int nblocks = rows / ROWS_PER_WG;         // 128
  xor_conv_wht_kernel<<<nblocks, NTHREADS, 0, stream>>>(p1, p2, out);
}
